// TensorReduction_55422257988202
// MI455X (gfx1250) — matrix operand order repaired and checked by reading, not compiled, not run on hardware
//
#include <hip/hip_runtime.h>

typedef __attribute__((ext_vector_type(16))) _Float16 v16h;
typedef __attribute__((ext_vector_type(4)))  _Float16 v4h;
typedef __attribute__((ext_vector_type(8)))  float    v8f;
typedef __attribute__((ext_vector_type(4)))  int      v4i;

#define N_ROWS 4096
#define DIM1   512
#define DIM2   128
#define BN     128              // n-rows per workgroup
#define ISPLIT 8                // split of the i (outer K) dimension across blocks
#define IBLK   (DIM1 / ISPLIT)  // 64 i-values per workgroup
#define NJC    (DIM1 / 32)      // 16 j-chunks of 32

#define HAS_ASYNC_LDS __has_builtin(__builtin_amdgcn_global_load_async_to_lds_b128)

__device__ __forceinline__ void wait_async_lds()
{
#if __has_builtin(__builtin_amdgcn_s_wait_asynccnt)
    __builtin_amdgcn_s_wait_asynccnt(0);
#elif HAS_ASYNC_LDS
    asm volatile("s_wait_asynccnt 0x0" ::: "memory");
#endif
}

// One-time W f32 -> f16 conversion (bandwidth-trivial vs the 275-GFLOP compute).
__global__ void trq_convw(const float* __restrict__ W, _Float16* __restrict__ Wh)
{
    const size_t idx = (size_t)blockIdx.x * blockDim.x + threadIdx.x; // 4 floats each
    const float4 f = ((const float4*)W)[idx];
    v4h h;
    h[0] = (_Float16)f.x; h[1] = (_Float16)f.y;
    h[2] = (_Float16)f.z; h[3] = (_Float16)f.w;
    ((v4h*)Wh)[idx] = h;
}

// Y[n,k] = sum_{i,j} X[n,i] * W[k,i,j] * X[n,j]
// GEMM view: A[n,(i,j)] = X[n,i]*X[n,j] (built on the fly, f16)
//            B[(i,j),k] = W[k,i,j]      (staged to LDS, shared by 8 waves)
// Wave tile: 2 row-strips x 4 k-tiles -> each B fragment feeds 2 WMMAs.
// WF16=1: B streamed from pre-converted f16 W via global_load_async_to_lds_b128.
// WF16=0: B converted f32->f16 through VGPRs while staging (no workspace needed).
template <int WF16>
__global__ __launch_bounds__(256)
void trq_main(const float* __restrict__ X, const float* __restrict__ W,
              const _Float16* __restrict__ Wh, float* __restrict__ out,
              float* __restrict__ ws, int use_ws)
{
    __shared__ float    Xi[IBLK][BN + 1];   // X[n-block, i-block], padded (33 KB)
    __shared__ _Float16 Wl[2][BN][32];      // double-buffered W tile, f16 (16 KB)

    const int tid  = threadIdx.x;
    const int lane = tid & 31;
    const int wv   = tid >> 5;
    const int hh   = (lane >> 4) & 1;       // wave half (K-offset selector)
    const int lm   = lane & 15;
    const int rs   = wv & 3;                // row group: rows [rs*32, rs*32+32)
    const int kg   = wv >> 2;               // k group: tiles [kg*4, kg*4+4)

    const int n0 = blockIdx.x * BN;         // 32 n-blocks
    const int i0 = blockIdx.y * IBLK;       // 8 i-splits

    // ---- stage X[n0:+128, i0:+64] -> LDS (transposed, padded) ----
    for (int idx = tid; idx < BN * IBLK; idx += 256) {
        int nl = idx >> 6;
        int il = idx & (IBLK - 1);
        Xi[il][nl] = X[(size_t)(n0 + nl) * DIM1 + (i0 + il)];
    }
    __syncthreads();

    const int    row0  = n0 + rs * 32 + lm;        // A rows for the two strips
    const int    row1  = row0 + 16;
    const float* Xrow0 = X + (size_t)row0 * DIM1;
    const float* Xrow1 = X + (size_t)row1 * DIM1;

    v8f acc0[4] = {};
    v8f acc1[4] = {};

    // cooperative W stager: Wl[buf][k][jj] = (f16) W[k, i, j0+jj], 128x32 tile
    auto stageW = [&](int buf, int i, int j0) {
        const int k  = tid >> 1;
        const int jh = (tid & 1) * 16;
        if constexpr (WF16) {
            const _Float16* src = Wh + (size_t)k * (DIM1 * DIM1)
                                     + (size_t)i * DIM1 + j0 + jh;
            _Float16* dst = &Wl[buf][k][jh];
#if HAS_ASYNC_LDS
            __builtin_amdgcn_global_load_async_to_lds_b128((v4i*)src,       (v4i*)dst,       0, 0);
            __builtin_amdgcn_global_load_async_to_lds_b128((v4i*)(src + 8), (v4i*)(dst + 8), 0, 0);
#else
            const uint4* s4 = (const uint4*)src;
            uint4*       d4 = (uint4*)dst;
            d4[0] = s4[0];
            d4[1] = s4[1];
#endif
            __builtin_prefetch((const void*)(src + 4 * DIM1), 0, 1);
        } else {
            const float4* wp4 = (const float4*)(W + (size_t)k * (DIM1 * DIM1)
                                                  + (size_t)i * DIM1 + j0 + jh);
            float4 a0 = wp4[0], a1 = wp4[1], a2 = wp4[2], a3 = wp4[3];
            __builtin_prefetch((const void*)(wp4 + 2 * (DIM1 / 4)), 0, 1);
            float f[16];
            f[0]=a0.x; f[1]=a0.y; f[2]=a0.z; f[3]=a0.w;
            f[4]=a1.x; f[5]=a1.y; f[6]=a1.z; f[7]=a1.w;
            f[8]=a2.x; f[9]=a2.y; f[10]=a2.z; f[11]=a2.w;
            f[12]=a3.x; f[13]=a3.y; f[14]=a3.z; f[15]=a3.w;
            v16h hv;
            #pragma unroll
            for (int e = 0; e < 16; ++e) hv[e] = (_Float16)f[e];
            *(v16h*)&Wl[buf][k][jh] = hv;
        }
    };

    for (int jc = 0; jc < NJC; ++jc) {
        const int j0 = jc * 32;

        // per-lane j-slices of X (one contiguous 16-run: k = 16*half + e)
        float xj0[16], xj1[16];
        {
            const float4* p0 = (const float4*)(Xrow0 + j0 + 16 * hh);
            const float4* p1 = (const float4*)(Xrow0 + j0 + 16 * hh + 8);
            float4 q0 = p0[0], q1 = p0[1], q2 = p1[0], q3 = p1[1];
            xj0[0]=q0.x; xj0[1]=q0.y; xj0[2]=q0.z; xj0[3]=q0.w;
            xj0[4]=q1.x; xj0[5]=q1.y; xj0[6]=q1.z; xj0[7]=q1.w;
            xj0[8]=q2.x; xj0[9]=q2.y; xj0[10]=q2.z; xj0[11]=q2.w;
            xj0[12]=q3.x; xj0[13]=q3.y; xj0[14]=q3.z; xj0[15]=q3.w;
            const float4* p2 = (const float4*)(Xrow1 + j0 + 16 * hh);
            const float4* p3 = (const float4*)(Xrow1 + j0 + 16 * hh + 8);
            float4 r0 = p2[0], r1 = p2[1], r2 = p3[0], r3 = p3[1];
            xj1[0]=r0.x; xj1[1]=r0.y; xj1[2]=r0.z; xj1[3]=r0.w;
            xj1[4]=r1.x; xj1[5]=r1.y; xj1[6]=r1.z; xj1[7]=r1.w;
            xj1[8]=r2.x; xj1[9]=r2.y; xj1[10]=r2.z; xj1[11]=r2.w;
            xj1[12]=r3.x; xj1[13]=r3.y; xj1[14]=r3.z; xj1[15]=r3.w;
        }

        stageW(0, i0, j0);
        if constexpr (WF16) wait_async_lds();
        __syncthreads();
        int cur = 0;

        for (int it = 0; it < IBLK; ++it) {
            const int nxt = cur ^ 1;
            if (it + 1 < IBLK) stageW(nxt, i0 + it + 1, j0);

            // A fragments: scalar X[row,i] times hoisted j-slice, cvt to f16
            const float s0 = Xi[it][rs * 32 + lm];
            const float s1 = Xi[it][rs * 32 + 16 + lm];
            v16h a0, a1;
            #pragma unroll
            for (int e = 0; e < 16; ++e) {
                a0[e] = (_Float16)(s0 * xj0[e]);
                a1[e] = (_Float16)(s1 * xj1[e]);
            }

            // 4 k-tiles, each B fragment feeds both row-strips (2 WMMAs)
            #pragma unroll
            for (int t = 0; t < 4; ++t) {
                const v16h b = *(const v16h*)&Wl[cur][(kg * 4 + t) * 16 + lm][hh * 16];
                acc0[t] = __builtin_amdgcn_wmma_f32_16x16x32_f16(
                    false, a0, false, b, (short)0, acc0[t], false, false);
                acc1[t] = __builtin_amdgcn_wmma_f32_16x16x32_f16(
                    false, a1, false, b, (short)0, acc1[t], false, false);
            }
            if constexpr (WF16) wait_async_lds();
            __syncthreads();
            cur = nxt;
        }
    }

    // ---- write partials (D layout: VGPR r -> M = r + 8*half, lane%16 -> N) ----
    if (use_ws) {
        float* wsp = ws + (size_t)blockIdx.y * ((size_t)N_ROWS * DIM2);
        #pragma unroll
        for (int t = 0; t < 4; ++t)
            #pragma unroll
            for (int r = 0; r < 8; ++r) {
                const int k = (kg * 4 + t) * 16 + lm;
                wsp[(size_t)(row0 - lm + r + 8 * hh) * DIM2 + k] = acc0[t][r];
                wsp[(size_t)(row1 - lm + r + 8 * hh) * DIM2 + k] = acc1[t][r];
            }
    } else {
        #pragma unroll
        for (int t = 0; t < 4; ++t)
            #pragma unroll
            for (int r = 0; r < 8; ++r) {
                const int k = (kg * 4 + t) * 16 + lm;
                atomicAdd(&out[(size_t)(row0 - lm + r + 8 * hh) * DIM2 + k], acc0[t][r]);
                atomicAdd(&out[(size_t)(row1 - lm + r + 8 * hh) * DIM2 + k], acc1[t][r]);
            }
    }
}

__global__ void trq_reduce(const float* __restrict__ ws, float* __restrict__ out)
{
    const int e = blockIdx.x * blockDim.x + threadIdx.x;
    if (e < N_ROWS * DIM2) {
        float s = 0.f;
        #pragma unroll
        for (int p = 0; p < ISPLIT; ++p)
            s += ws[(size_t)p * ((size_t)N_ROWS * DIM2) + e];
        out[e] = s;
    }
}

__global__ void trq_zero(float* __restrict__ out)
{
    const int e = blockIdx.x * blockDim.x + threadIdx.x;
    if (e < N_ROWS * DIM2) out[e] = 0.f;
}

extern "C" void kernel_launch(void* const* d_in, const int* in_sizes, int n_in,
                              void* d_out, int out_size, void* d_ws, size_t ws_size,
                              hipStream_t stream)
{
    const float* X = (const float*)d_in[0];   // [4096, 512] f32
    const float* W = (const float*)d_in[1];   // [128, 512, 512] f32
    float* out = (float*)d_out;               // [4096, 128] f32

    const size_t partialBytes = (size_t)ISPLIT * N_ROWS * DIM2 * sizeof(float); // 16 MB
    const size_t whBytes      = (size_t)DIM2 * DIM1 * DIM1 * sizeof(_Float16);  // 64 MB

    const int use_ws = (d_ws != nullptr && ws_size >= partialBytes) ? 1 : 0;
    const int w_f16  = (d_ws != nullptr && ws_size >= partialBytes + whBytes) ? 1 : 0;

    float*     ws = (float*)d_ws;
    _Float16*  Wh = w_f16 ? (_Float16*)((char*)d_ws + partialBytes) : nullptr;

    const int outElems = N_ROWS * DIM2;
    dim3 grid(N_ROWS / BN, ISPLIT);

    if (w_f16) {
        const int nVec4 = DIM2 * DIM1 * DIM1 / 4;   // 8,388,608
        trq_convw<<<nVec4 / 256, 256, 0, stream>>>(W, Wh);
    }

    if (!use_ws)
        trq_zero<<<(outElems + 255) / 256, 256, 0, stream>>>(out);

    if (w_f16)
        trq_main<1><<<grid, 256, 0, stream>>>(X, W, Wh, out, ws, use_ws);
    else
        trq_main<0><<<grid, 256, 0, stream>>>(X, W, Wh, out, ws, use_ws);

    if (use_ws)
        trq_reduce<<<(outElems + 255) / 256, 256, 0, stream>>>(ws, out);
}
